// BigST_35837207118640
// MI455X (gfx1250) — matrix operand order repaired and checked by reading, not compiled, not run on hardware
//
#include <hip/hip_runtime.h>
#include <math.h>

#define HIDD 32
#define H4   128
#define RF   64
#define TOUT 12
#define NN   50000
#define BBATCH 4
#define CHUNK 256

typedef __attribute__((ext_vector_type(16))) _Float16 v16h;
typedef __attribute__((ext_vector_type(8)))  float    v8f;

static __device__ __forceinline__ v8f wmma16(v16h a, v16h b, v8f c) {
  // D = A(16x32 f16) * B(32x16 f16) + C(16x16 f32)
  return __builtin_amdgcn_wmma_f32_16x16x32_f16(false, a, false, b, (short)0, c, false, false);
}

// A-fragment from an f16 row: halfs j -> K = kb*32 + 16*g + j
// One contiguous 32-byte run -> global_load_b128 pairs, zero cvts.
static __device__ __forceinline__ v16h loadA_f16row(const _Float16* __restrict__ row, int kb, int g) {
  v16h a;
  const _Float16* p0 = row + kb * 32 + 16 * g;
  const _Float16* p1 = p0 + 8;
#pragma unroll
  for (int t = 0; t < 8; ++t) { a[t] = p0[t]; a[8 + t] = p1[t]; }
  return a;
}

// B-fragment, contiguous f16: halfs j -> K = base + 16*g + j (caller bakes in col*ld + kb*32 + 16*g)
static __device__ __forceinline__ v16h loadB_f16(const _Float16* __restrict__ p) {
  v16h b;
#pragma unroll
  for (int t = 0; t < 16; ++t) b[t] = p[t];
  return b;
}

// order-preserving float<->uint key for atomic max
static __device__ __forceinline__ unsigned fkey(float x) {
  unsigned u = __float_as_uint(x);
  return (u & 0x80000000u) ? ~u : (u | 0x80000000u);
}
static __device__ __forceinline__ float funkey(unsigned k) {
  return __uint_as_float((k & 0x80000000u) ? (k & 0x7fffffffu) : ~k);
}

// ---------------------------------------------------------------- K0: prep
// f16-convert GLU weights (4x 128x128) and padded regression weights (16x256), init kmax keys.
__global__ void k0_prep(const float* __restrict__ wi0, const float* __restrict__ wo0,
                        const float* __restrict__ wi1, const float* __restrict__ wo1,
                        const float* __restrict__ wreg,
                        _Float16* __restrict__ w16, _Float16* __restrict__ wreg16,
                        unsigned* __restrict__ kmax) {
  int idx = blockIdx.x * blockDim.x + threadIdx.x;
  if (idx < 4 * H4 * H4) {
    int m = idx >> 14, r = idx & (H4 * H4 - 1);
    const float* src = (m == 0) ? wi0 : (m == 1) ? wo0 : (m == 2) ? wi1 : wo1;
    w16[idx] = (_Float16)src[r];
  }
  int i2 = idx - 4 * H4 * H4;
  if (i2 >= 0 && i2 < 16 * 256) {
    int t = i2 >> 8, i = i2 & 255;
    wreg16[i2] = (_Float16)((t < TOUT) ? wreg[t * 256 + i] : 0.f);
  }
  if (idx < BBATCH) kmax[idx] = 0u;
}

// ---------------------------------------------------------------- K1: embed + Performer features
// One thread per node. Produces h0,h [B,N,128] (f16), q [B,N,64] (f16), (dash_k - diag) in kbuf (f16),
// and global per-batch max of dash_k via key-atomicMax. All math in f32, storage f16.
__global__ __launch_bounds__(64) void k1_embed(
    const float* __restrict__ x, const float* __restrict__ node_emb,
    const float* __restrict__ time_emb, const float* __restrict__ week_emb,
    const float* __restrict__ W_in, const float* __restrict__ b_in,
    const float* __restrict__ W1, const float* __restrict__ b1,
    const float* __restrict__ W2, const float* __restrict__ b2,
    const float* __restrict__ proj,
    _Float16* __restrict__ h0, _Float16* __restrict__ h,
    _Float16* __restrict__ q, _Float16* __restrict__ kbuf,
    unsigned* __restrict__ kmax) {
  __shared__ float sWin[32 * 36], sW1[32 * 96], sW2[32 * 96], sproj[64 * 32];
  __shared__ float sb_in[32], sb1[32], sb2[32];
  __shared__ float sxg[64 * 96];
  __shared__ unsigned red;
  const int tid = threadIdx.x;
  for (int i = tid; i < 32 * 36; i += 64) sWin[i] = W_in[i];
  for (int i = tid; i < 32 * 96; i += 64) { sW1[i] = W1[i]; sW2[i] = W2[i]; }
  for (int i = tid; i < 64 * 32; i += 64) sproj[i] = proj[i];
  if (tid < 32) { sb_in[tid] = b_in[tid]; sb1[tid] = b1[tid]; sb2[tid] = b2[tid]; }
  if (tid == 0) red = 0u;
  __syncthreads();

  const int b = blockIdx.y;
  int n = blockIdx.x * 64 + tid;
  const bool valid = n < NN;
  const int nc = valid ? n : NN - 1;
  const long bn = (long)b * NN + nc;
  const float* xr = x + bn * 36;
  _Float16* h0r = h0 + bn * H4;
  _Float16* hr  = h  + bn * H4;
  float* sg = sxg + tid * 96;

  int ti = (int)(xr[34] * 288.f); ti = ti < 0 ? 0 : (ti > 287 ? 287 : ti);
  int wk = (int)(xr[35]);         wk = wk < 0 ? 0 : (wk > 6 ? 6 : wk);
  for (int i = 0; i < 32; ++i) {
    sg[i]      = node_emb[nc * 32 + i];
    sg[32 + i] = time_emb[ti * 32 + i];
    sg[64 + i] = week_emb[wk * 32 + i];
  }
  // inp = x(36) @ W_in.T + b_in ; h = [inp, ne, te, we]
  for (int o = 0; o < 32; ++o) {
    float acc = sb_in[o];
    for (int j = 0; j < 36; ++j) acc += xr[j] * sWin[o * 36 + j];
    if (valid) { h0r[o] = (_Float16)acc; hr[o] = (_Float16)acc; }
  }
  if (valid)
    for (int i = 0; i < 96; ++i) {
      h0r[32 + i] = (_Float16)sg[i];
      hr[32 + i]  = (_Float16)sg[i];
    }

  const float dsc = 0.42044820762685725f;  // 32^-0.25 (tau = 1)
  float nv[32];

  // ---- q features ----
#pragma unroll
  for (int o = 0; o < 32; ++o) {
    float acc = sb1[o];
    for (int i = 0; i < 96; ++i) acc += sg[i] * sW1[o * 96 + i];
    nv[o] = acc * dsc;
  }
  float diag = 0.f;
#pragma unroll
  for (int o = 0; o < 32; ++o) diag += nv[o] * nv[o];
  diag *= 0.5f;
  _Float16* qrow = q + bn * RF;
  float mx = -3.0e38f;
  for (int m = 0; m < RF; ++m) {
    float d = 0.f;
#pragma unroll
    for (int i = 0; i < 32; ++i) d += nv[i] * sproj[m * 32 + i];
    if (valid) qrow[m] = (_Float16)d;  // stash raw dash
    mx = fmaxf(mx, d);
  }
  if (valid)
    for (int m = 0; m < RF; ++m) {
      float dv = (float)qrow[m];
      qrow[m] = (_Float16)(0.125f * (__expf(dv - diag - mx) + 1e-6f));
    }

  // ---- k features (exp deferred to K2; needs global max of dash) ----
#pragma unroll
  for (int o = 0; o < 32; ++o) {
    float acc = sb2[o];
    for (int i = 0; i < 96; ++i) acc += sg[i] * sW2[o * 96 + i];
    nv[o] = acc * dsc;
  }
  diag = 0.f;
#pragma unroll
  for (int o = 0; o < 32; ++o) diag += nv[o] * nv[o];
  diag *= 0.5f;
  _Float16* krow = kbuf + bn * RF;
  float mxk = -3.0e38f;
  for (int m = 0; m < RF; ++m) {
    float d = 0.f;
#pragma unroll
    for (int i = 0; i < 32; ++i) d += nv[i] * sproj[m * 32 + i];
    if (valid) krow[m] = (_Float16)(d - diag);
    mxk = fmaxf(mxk, d);
  }
  atomicMax(&red, fkey(mxk));
  __syncthreads();
  if (tid == 0) atomicMax(&kmax[b], red);
}

// ---------------------------------------------------------------- K2: finalize k with global max
__global__ void k2_fin(const unsigned* __restrict__ kmax, _Float16* __restrict__ kbuf) {
  const int b = blockIdx.y;
  long idx = (long)blockIdx.x * blockDim.x + threadIdx.x;
  if (idx < (long)NN * RF) {
    float mx = funkey(kmax[b]);
    long o = (long)b * NN * RF + idx;
    float vv = (float)kbuf[o];
    kbuf[o] = (_Float16)(0.125f * (__expf(vv - mx) + 1e-6f));
  }
}

// ---------------------------------------------------------------- K_L1: GLU(h) via WMMA, kv += k^T v via WMMA
// 128 threads = 4 waves. Wave w owns d-columns [32w,32w+32). Block streams CHUNK nodes of one batch.
__global__ __launch_bounds__(128) void kl1(
    const _Float16* __restrict__ h, const _Float16* __restrict__ kq,
    const _Float16* __restrict__ wi16, const _Float16* __restrict__ wo16,
    const float* __restrict__ bi, const float* __restrict__ bo,
    float* __restrict__ kv, float* __restrict__ ksum) {
  __shared__ _Float16 lds_v[32 * H4];  // 8 KB: v (f16) for 32 nodes x 128 cols
  const int b = blockIdx.y;
  const int base0 = blockIdx.x * CHUNK;
  const int tid = threadIdx.x;
  const int w = tid >> 5, lane = tid & 31, nloc = lane & 15, g = lane >> 4;
  const long bOff = (long)b * NN;

  v8f zero = {};
  v8f kvC[4][2];
#pragma unroll
  for (int mt = 0; mt < 4; ++mt)
#pragma unroll
    for (int ds = 0; ds < 2; ++ds) kvC[mt][ds] = zero;
  float ksacc = 0.f;

  float bib[2], bob[2];
#pragma unroll
  for (int dt = 0; dt < 2; ++dt) {
    int o = 32 * w + 16 * dt + nloc;
    bib[dt] = bi[o]; bob[dt] = bo[o];
  }

  for (int s = 0; s < CHUNK; s += 32) {
    const int n0 = base0 + s;          // uniform across block
    if (n0 >= NN) break;
    {  // prefetch next step's h rows
      int np = n0 + 32 + nloc; np = np < NN ? np : NN - 1;
      __builtin_prefetch(h + (bOff + np) * H4, 0, 1);
    }
    // ---- stage 1: v = sigmoid(h@Wi^T+bi) * (h@Wo^T+bo), cols [32w,32w+32) ----
    v8f cwi[2][2], cwo[2][2];
#pragma unroll
    for (int st = 0; st < 2; ++st)
#pragma unroll
      for (int dt = 0; dt < 2; ++dt) { cwi[st][dt] = zero; cwo[st][dt] = zero; }
#pragma unroll
    for (int kb = 0; kb < 4; ++kb) {
      int na = n0 + nloc;      int nac = na < NN ? na : NN - 1;
      int nb = n0 + 16 + nloc; int nbc = nb < NN ? nb : NN - 1;
      v16h a0 = loadA_f16row(h + (bOff + nac) * H4, kb, g);
      v16h a1 = loadA_f16row(h + (bOff + nbc) * H4, kb, g);
#pragma unroll
      for (int dt = 0; dt < 2; ++dt) {
        int o = 32 * w + 16 * dt + nloc;
        v16h bwi = loadB_f16(wi16 + o * H4 + kb * 32 + 16 * g);
        v16h bwo = loadB_f16(wo16 + o * H4 + kb * 32 + 16 * g);
        cwi[0][dt] = wmma16(a0, bwi, cwi[0][dt]);
        cwi[1][dt] = wmma16(a1, bwi, cwi[1][dt]);
        cwo[0][dt] = wmma16(a0, bwo, cwo[0][dt]);
        cwo[1][dt] = wmma16(a1, bwo, cwo[1][dt]);
      }
    }
#pragma unroll
    for (int st = 0; st < 2; ++st)
#pragma unroll
      for (int dt = 0; dt < 2; ++dt) {
        int dcol = 32 * w + 16 * dt + nloc;
#pragma unroll
        for (int r = 0; r < 8; ++r) {
          float gate = cwi[st][dt][r] + bib[dt];
          float lin  = cwo[st][dt][r] + bob[dt];
          lds_v[(st * 16 + r + 8 * g) * H4 + dcol] = (_Float16)(lin / (1.f + __expf(-gate)));
        }
      }
    __syncthreads();
    // ---- stage 2: kvC += k^T(64 x 32nodes) * v(32nodes x [32w..32w+32)) ----
#pragma unroll
    for (int mt = 0; mt < 4; ++mt) {
      int m = mt * 16 + nloc;
      v16h ak;
#pragma unroll
      for (int jj = 0; jj < 16; ++jj) {
        int noff = 16 * g + jj;
        int nn2 = n0 + noff;
        ak[jj] = (nn2 < NN) ? kq[(bOff + nn2) * RF + m] : (_Float16)0.f;
      }
#pragma unroll
      for (int ds = 0; ds < 2; ++ds) {
        int dcol = 32 * w + ds * 16 + nloc;
        v16h bv;
#pragma unroll
        for (int jj = 0; jj < 16; ++jj)
          bv[jj] = lds_v[(16 * g + jj) * H4 + dcol];
        kvC[mt][ds] = wmma16(ak, bv, kvC[mt][ds]);
      }
    }
    if (tid < RF) {
      for (int no = 0; no < 32; ++no) {
        int nn2 = n0 + no;
        if (nn2 < NN) ksacc += (float)kq[(bOff + nn2) * RF + tid];
      }
    }
    __syncthreads();
  }
  // flush partials
#pragma unroll
  for (int mt = 0; mt < 4; ++mt)
#pragma unroll
    for (int ds = 0; ds < 2; ++ds) {
      int dcol = 32 * w + ds * 16 + nloc;
#pragma unroll
      for (int r = 0; r < 8; ++r) {
        int m = mt * 16 + r + 8 * g;
        atomicAdd(&kv[(b * RF + m) * H4 + dcol], kvC[mt][ds][r]);
      }
    }
  if (tid < RF) atomicAdd(&ksum[b * RF + tid], ksacc);
}

// ---------------------------------------------------------------- K_L2: h = LN(q@kv / q@ksum + h)
__global__ __launch_bounds__(128) void kl2(
    _Float16* __restrict__ h, const _Float16* __restrict__ q,
    const float* __restrict__ kv, const float* __restrict__ ksum,
    const float* __restrict__ lng, const float* __restrict__ lnb) {
  __shared__ _Float16 lds_kv[RF * H4];  // 16 KB
  __shared__ float lds_ks[RF];
  __shared__ float lds_out[64 * H4];    // 32 KB
  const int b = blockIdx.y;
  const int nb = blockIdx.x * 64;
  const int tid = threadIdx.x;
  const int w = tid >> 5, lane = tid & 31, nloc = lane & 15, g = lane >> 4;
  const long bOff = (long)b * NN;

  for (int i = tid; i < RF * H4; i += 128) lds_kv[i] = (_Float16)kv[b * RF * H4 + i];
  if (tid < RF) lds_ks[tid] = ksum[b * RF + tid];
  __syncthreads();

  const int tb = nb + 16 * w;
  int na = tb + nloc; int nac = na < NN ? na : NN - 1;
  const _Float16* qrow = q + (bOff + nac) * RF;
  v16h aq0 = loadA_f16row(qrow, 0, g);
  v16h aq1 = loadA_f16row(qrow, 1, g);
#pragma unroll
  for (int db = 0; db < 8; ++db) {
    v8f c = {};
    v16h b0, b1v;
#pragma unroll
    for (int jj = 0; jj < 16; ++jj) {
      b0[jj]  = lds_kv[(16 * g + jj) * H4 + db * 16 + nloc];
      b1v[jj] = lds_kv[(32 + 16 * g + jj) * H4 + db * 16 + nloc];
    }
    c = wmma16(aq0, b0, c);
    c = wmma16(aq1, b1v, c);
#pragma unroll
    for (int r = 0; r < 8; ++r)
      lds_out[(16 * w + r + 8 * g) * H4 + db * 16 + nloc] = c[r];
  }
  __syncthreads();

  if (tid < 64) {
    int n = nb + tid;
    if (n < NN) {
      const _Float16* qr = q + (bOff + n) * RF;
      float o2 = 0.f;
#pragma unroll
      for (int m = 0; m < RF; ++m) o2 += (float)qr[m] * lds_ks[m];
      float inv = 1.f / o2;
      _Float16* hr = h + (bOff + n) * H4;
      float s1 = 0.f, s2 = 0.f;
      for (int d = 0; d < H4; ++d) {
        float val = lds_out[tid * H4 + d] * inv + (float)hr[d];
        lds_out[tid * H4 + d] = val;
        s1 += val; s2 += val * val;
      }
      float mu = s1 * (1.f / H4);
      float var = s2 * (1.f / H4) - mu * mu;
      float rstd = rsqrtf(var + 1e-5f);
      for (int d = 0; d < H4; ++d)
        hr[d] = (_Float16)((lds_out[tid * H4 + d] - mu) * rstd * lng[d] + lnb[d]);
    }
  }
}

// ---------------------------------------------------------------- K3: out[b,t,n] = relu([h0,h]) @ Wreg^T + breg
__global__ __launch_bounds__(128) void k3_out(
    const _Float16* __restrict__ h0, const _Float16* __restrict__ h,
    const _Float16* __restrict__ wreg16, const float* __restrict__ breg,
    float* __restrict__ out) {
  __shared__ _Float16 lds_w[16 * 256];  // 8 KB, t-padded to 16
  const int b = blockIdx.y;
  const int nb = blockIdx.x * 64;
  const int tid = threadIdx.x;
  const int w = tid >> 5, lane = tid & 31, nloc = lane & 15, g = lane >> 4;
  const long bOff = (long)b * NN;
  for (int i = tid; i < 16 * 256; i += 128) lds_w[i] = wreg16[i];
  __syncthreads();

  const int tb = nb + 16 * w;
  int na = tb + nloc; int nac = na < NN ? na : NN - 1;
  const _Float16* r0 = h0 + (bOff + nac) * H4;
  const _Float16* r1 = h  + (bOff + nac) * H4;
  v8f c = {};
#pragma unroll
  for (int kb = 0; kb < 8; ++kb) {
    v16h a;
#pragma unroll
    for (int jj = 0; jj < 16; ++jj) {
      int i = kb * 32 + 16 * g + jj;
      _Float16 v = (i < H4) ? r0[i] : r1[i - H4];
      a[jj] = (v > (_Float16)0.f) ? v : (_Float16)0.f;
    }
    v16h bw;
#pragma unroll
    for (int jj = 0; jj < 16; ++jj)
      bw[jj] = lds_w[nloc * 256 + kb * 32 + 16 * g + jj];
    c = wmma16(a, bw, c);
  }
  if (nloc < TOUT) {
    float bt = breg[nloc];
#pragma unroll
    for (int r = 0; r < 8; ++r) {
      int n = tb + r + 8 * g;
      if (n < NN) out[((long)(b * TOUT + nloc)) * NN + n] = c[r] + bt;
    }
  }
}

// ---------------------------------------------------------------- launch
extern "C" void kernel_launch(void* const* d_in, const int* in_sizes, int n_in,
                              void* d_out, int out_size, void* d_ws, size_t ws_size,
                              hipStream_t stream) {
  const float* x        = (const float*)d_in[0];
  const float* node_emb = (const float*)d_in[1];
  const float* time_emb = (const float*)d_in[2];
  const float* week_emb = (const float*)d_in[3];
  const float* W_in     = (const float*)d_in[4];
  const float* b_in     = (const float*)d_in[5];
  const float* W1       = (const float*)d_in[6];
  const float* b1       = (const float*)d_in[7];
  const float* W2       = (const float*)d_in[8];
  const float* b2       = (const float*)d_in[9];
  const float* W_reg    = (const float*)d_in[10];
  const float* b_reg    = (const float*)d_in[11];
  const float* proj     = (const float*)d_in[12];

  char* ws = (char*)d_ws;
  size_t off = 0;
  auto carve = [&](size_t bytes) { void* p = ws + off; off += (bytes + 255) & ~(size_t)255; return p; };
  _Float16* h0     = (_Float16*)carve((size_t)BBATCH * NN * H4 * 2);
  _Float16* h      = (_Float16*)carve((size_t)BBATCH * NN * H4 * 2);
  _Float16* q      = (_Float16*)carve((size_t)BBATCH * NN * RF * 2);
  _Float16* kbuf   = (_Float16*)carve((size_t)BBATCH * NN * RF * 2);
  float*    kv     = (float*)carve((size_t)BBATCH * RF * H4 * 4);
  float*    ksum   = (float*)carve((size_t)BBATCH * RF * 4);
  _Float16* w16    = (_Float16*)carve((size_t)4 * H4 * H4 * 2);
  _Float16* wreg16 = (_Float16*)carve((size_t)16 * 256 * 2);
  unsigned* kmax   = (unsigned*)carve((size_t)BBATCH * 4);

  // K0: weight conversion + kmax key init
  {
    int total = 4 * H4 * H4 + 16 * 256;
    k0_prep<<<(total + 255) / 256, 256, 0, stream>>>(
        (const float*)d_in[13], (const float*)d_in[15],
        (const float*)d_in[19], (const float*)d_in[21],
        W_reg, w16, wreg16, kmax);
  }
  // K1: embeddings + Performer features
  {
    dim3 grid((NN + 63) / 64, BBATCH);
    k1_embed<<<grid, 64, 0, stream>>>(x, node_emb, time_emb, week_emb,
                                      W_in, b_in, W1, b1, W2, b2, proj,
                                      h0, h, q, kbuf, kmax);
  }
  // K2: finalize k with global max
  {
    dim3 grid(((long)NN * RF + 255) / 256, BBATCH);
    k2_fin<<<grid, 256, 0, stream>>>(kmax, kbuf);
  }
  // Two Performer-GLU layers
  for (int L = 0; L < 2; ++L) {
    hipMemsetAsync(kv, 0, (size_t)BBATCH * RF * H4 * 4, stream);
    hipMemsetAsync(ksum, 0, (size_t)BBATCH * RF * 4, stream);
    const _Float16* wi16 = w16 + (size_t)(2 * L) * H4 * H4;
    const _Float16* wo16 = w16 + (size_t)(2 * L + 1) * H4 * H4;
    const float* bi = (const float*)d_in[14 + 6 * L];
    const float* bo = (const float*)d_in[16 + 6 * L];
    const float* lg = (const float*)d_in[17 + 6 * L];
    const float* lb = (const float*)d_in[18 + 6 * L];
    dim3 g1((NN + CHUNK - 1) / CHUNK, BBATCH);
    kl1<<<g1, 128, 0, stream>>>(h, kbuf, wi16, wo16, bi, bo, kv, ksum);
    dim3 g2((NN + 63) / 64, BBATCH);
    kl2<<<g2, 128, 0, stream>>>(h, q, kv, ksum, lg, lb);
  }
  // K3: regression head
  {
    dim3 grid((NN + 63) / 64, BBATCH);
    k3_out<<<grid, 128, 0, stream>>>(h0, h, wreg16, b_reg, (float*)d_out);
  }
}
